// NeRFGraph_21586505629820
// MI455X (gfx1250) — matrix operand order repaired and checked by reading, not compiled, not run on hardware
//
#include <hip/hip_runtime.h>
#include <hip/hip_bf16.h>

typedef _Float16 v16h __attribute__((ext_vector_type(16)));
typedef _Float16 v8h  __attribute__((ext_vector_type(8)));
typedef float    v8f  __attribute__((ext_vector_type(8)));
typedef float    v4f  __attribute__((ext_vector_type(4)));

constexpr int NB   = 40960;      // batch
constexpr int NB3  = NB * 3;     // edges
constexpr int NG   = 20;         // groups
constexpr int GSZ  = NB / NG;    // 2048 per group
constexpr int XDIM = 90;         // 63 xyz + 27 dir

// ---------------------------------------------------------------------------
// Generic WMMA GEMM: C[M x N](f16) = act(A[M x K](f16) @ W[K x N](f16) + bias)
// W is provided PRE-TRANSPOSED as WT[N x K] so both LDS tiles stage with
// b128 vector copies (no in-loop transpose).
// Block tile 128x128, 8 waves (4 in M, 2 in N), wave tile 32x64.
// Requires: M % 128 == 0, N % 128 == 0, K % 32 == 0.
// ---------------------------------------------------------------------------
template<bool RELU, bool HAS_BIAS>
__global__ __launch_bounds__(256)
void wmma_gemm(const _Float16* __restrict__ A, const _Float16* __restrict__ WT,
               const float* __restrict__ bias, _Float16* __restrict__ C,
               int K, int N, int ldc)
{
    __shared__ _Float16 As[128 * 32];   // [m][k]
    __shared__ _Float16 Ws[128 * 32];   // [n][k]

    const int tid  = threadIdx.x;
    const int lane = tid & 31;
    const int wave = tid >> 5;
    const int wm   = wave & 3;          // 0..3 (M dir)
    const int wn   = wave >> 2;         // 0..1 (N dir)
    const int m0   = blockIdx.y * 128;
    const int n0   = blockIdx.x * 128;

    const v8f zero = {};
    v8f acc[2][4];
#pragma unroll
    for (int i = 0; i < 2; i++)
#pragma unroll
        for (int j = 0; j < 4; j++) acc[i][j] = zero;

    const int srow = tid >> 1, shalf = tid & 1;   // both tiles: 128 rows x 32

    for (int kt = 0; kt < K; kt += 32) {
        // ---- stage A tile (row-major) ----
        {
            const _Float16* src = A + (size_t)(m0 + srow) * K + kt + shalf * 16;
            v8h a0 = *(const v8h*)(src);
            v8h a1 = *(const v8h*)(src + 8);
            *(v8h*)&As[srow * 32 + shalf * 16]     = a0;
            *(v8h*)&As[srow * 32 + shalf * 16 + 8] = a1;
            if (kt + 32 < K) __builtin_prefetch(src + 32, 0, 1);
        }
        // ---- stage W tile (WT is [n][k], direct vector copy) ----
        {
            const _Float16* src = WT + (size_t)(n0 + srow) * K + kt + shalf * 16;
            v8h w0 = *(const v8h*)(src);
            v8h w1 = *(const v8h*)(src + 8);
            *(v8h*)&Ws[srow * 32 + shalf * 16]     = w0;
            *(v8h*)&Ws[srow * 32 + shalf * 16 + 8] = w1;
        }
        __syncthreads();

        // ---- build fragments per CDNA5 16-bit layouts ----
        const int hs = lane >> 4;       // half-select
        const int r  = lane & 15;
        v16h af[2], bf[4];
#pragma unroll
        for (int i = 0; i < 2; i++) {
            int row = wm * 32 + i * 16 + r;
            v8h lo = *(const v8h*)&As[row * 32 + hs * 8];        // K 0..7 / 8..15
            v8h hi = *(const v8h*)&As[row * 32 + 16 + hs * 8];   // K 16..23 / 24..31
#pragma unroll
            for (int e = 0; e < 8; e++) { af[i][e] = lo[e]; af[i][8 + e] = hi[e]; }
        }
#pragma unroll
        for (int j = 0; j < 4; j++) {
            int col = wn * 64 + j * 16 + r;
            v8h lo = *(const v8h*)&Ws[col * 32 + hs * 8];        // K 0..7 / 8..15
            v8h hi = *(const v8h*)&Ws[col * 32 + 16 + hs * 8];   // K 16..23 / 24..31
#pragma unroll
            for (int e = 0; e < 8; e++) { bf[j][e] = lo[e]; bf[j][8 + e] = hi[e]; }
        }
        // ---- 8 WMMAs per wave per K-step ----
#pragma unroll
        for (int i = 0; i < 2; i++)
#pragma unroll
            for (int j = 0; j < 4; j++)
                acc[i][j] = __builtin_amdgcn_wmma_f32_16x16x32_f16(
                    false, af[i], false, bf[j], (short)0, acc[i][j], false, false);
        __syncthreads();
    }

    // ---- epilogue: C/D layout -> VGPR r is row r (+8 for lanes 16..31) ----
    const int rh = (lane >> 4) * 8;
    const int cl = lane & 15;
#pragma unroll
    for (int i = 0; i < 2; i++) {
#pragma unroll
        for (int j = 0; j < 4; j++) {
            int gcol = n0 + wn * 64 + j * 16 + cl;
            float bv = 0.f;
            if (HAS_BIAS) bv = bias[gcol];
#pragma unroll
            for (int rr = 0; rr < 8; rr++) {
                int grow = m0 + wm * 32 + i * 16 + rh + rr;
                float v = acc[i][j][rr] + bv;
                if (RELU) v = v > 0.f ? v : 0.f;
                C[(size_t)grow * ldc + gcol] = (_Float16)v;
            }
        }
    }
}

// ---------------------------------------------------------------------------
// Group construction + kNN (K=3) within each group of 2048 points (63-D)
// ---------------------------------------------------------------------------
__global__ void zero_cnt(int* cnt) { if (threadIdx.x < 32) cnt[threadIdx.x] = 0; }

__global__ __launch_bounds__(256)
void build_groups(const int* __restrict__ bid, int* __restrict__ cnt, int* __restrict__ glist)
{
    int i = blockIdx.x * 256 + threadIdx.x;
    if (i >= NB) return;
    int g = bid[i];
    int slot = atomicAdd(&cnt[g], 1);
    glist[g * GSZ + slot] = i;
}

__global__ __launch_bounds__(256)
void knn_kernel(const float* __restrict__ x, const int* __restrict__ glist, int* __restrict__ nbr)
{
    __shared__ float sx[128 * 64];
    __shared__ int   sid[128];
    const int g    = blockIdx.x >> 3;       // 8 tiles of 256 queries per group
    const int tile = blockIdx.x & 7;
    const int tid  = threadIdx.x;
    const int qidx = glist[g * GSZ + tile * 256 + tid];

    float q[64];
#pragma unroll
    for (int k = 0; k < 64; k++) q[k] = (k < 63) ? x[(size_t)qidx * XDIM + k] : 0.f;

    float bd0 = 3.4e38f, bd1 = 3.4e38f, bd2 = 3.4e38f;
    int   bi0 = -1, bi1 = -1, bi2 = -1;

    for (int cb = 0; cb < GSZ; cb += 128) {
        const int cand = tid >> 1, h = tid & 1;
        const int cidx = glist[g * GSZ + cb + cand];
        if (h == 0) sid[cand] = cidx;
#pragma unroll
        for (int e = 0; e < 32; e++) {
            int col = h * 32 + e;
            sx[cand * 64 + col] = (col < 63) ? x[(size_t)cidx * XDIM + col] : 0.f;
        }
        __syncthreads();
        for (int c = 0; c < 128; c++) {
            int cid = sid[c];
            float d2 = 0.f;
#pragma unroll
            for (int k = 0; k < 64; k += 4) {
                v4f s = *(const v4f*)&sx[c * 64 + k];
                float a0 = q[k] - s[0], a1 = q[k + 1] - s[1];
                float a2 = q[k + 2] - s[2], a3 = q[k + 3] - s[3];
                d2 += a0 * a0 + a1 * a1 + a2 * a2 + a3 * a3;
            }
            if (cid != qidx && d2 < bd2) {
                if (d2 < bd0)      { bd2 = bd1; bi2 = bi1; bd1 = bd0; bi1 = bi0; bd0 = d2; bi0 = cid; }
                else if (d2 < bd1) { bd2 = bd1; bi2 = bi1; bd1 = d2;  bi1 = cid; }
                else               { bd2 = d2;  bi2 = cid; }
            }
        }
        __syncthreads();
    }
    nbr[qidx * 3 + 0] = bi0;
    nbr[qidx * 3 + 1] = bi1;
    nbr[qidx * 3 + 2] = bi2;
}

// ---------------------------------------------------------------------------
// Input staging + weight conversion (weights are stored TRANSPOSED: [N][K])
// ---------------------------------------------------------------------------
__global__ __launch_bounds__(256)
void prep_kernel(const float* __restrict__ x, _Float16* __restrict__ xyz16,
                 _Float16* __restrict__ cat320, _Float16* __restrict__ featdir)
{
    int i = blockIdx.x * 256 + threadIdx.x;
    if (i >= NB) return;
    const float* xr = x + (size_t)i * XDIM;
    for (int c = 0; c < 64; c++) {
        float v = (c < 63) ? xr[c] : 0.f;
        xyz16[(size_t)i * 64 + c] = (_Float16)v;
    }
    for (int c = 0; c < 63; c++) cat320[(size_t)i * 320 + c] = (_Float16)xr[c];
    cat320[(size_t)i * 320 + 319] = (_Float16)0.f;             // pad col
    for (int c = 0; c < 32; c++) {                             // dir + pad cols 256..287
        float v = (c < 27) ? xr[63 + c] : 0.f;
        featdir[(size_t)i * 288 + 256 + c] = (_Float16)v;
    }
}

// dst[n*Kdst + k] = (k < Ksrc) ? src[k*N + n] : 0   (f32 -> f16, transpose + pad)
__global__ __launch_bounds__(256)
void conv_pad_t(const float* __restrict__ src, _Float16* __restrict__ dst,
                int Ksrc, int Kdst, int N)
{
    int idx = blockIdx.x * 256 + threadIdx.x;
    if (idx >= Kdst * N) return;
    int n = idx / Kdst, k = idx - n * Kdst;
    float v = (k < Ksrc) ? src[(size_t)k * N + n] : 0.f;
    dst[idx] = (_Float16)v;
}

// split [Wtop; Wbot] into (Wtop - Wbot) and Wbot, transposed+padded to [N][Kdst]
__global__ __launch_bounds__(256)
void make_edge_w_t(const float* __restrict__ src, _Float16* __restrict__ dD,
                   _Float16* __restrict__ dB, int Khalf, int Kdst, int N)
{
    int idx = blockIdx.x * 256 + threadIdx.x;
    if (idx >= Kdst * N) return;
    int n = idx / Kdst, k = idx - n * Kdst;
    float t = 0.f, b = 0.f;
    if (k < Khalf) { t = src[(size_t)k * N + n]; b = src[(size_t)(k + Khalf) * N + n]; }
    dD[idx] = (_Float16)(t - b);
    dB[idx] = (_Float16)b;
}

// ---------------------------------------------------------------------------
// EdgeConv glue: per-edge relu(a_i + c_j), then mean over K=3
// ---------------------------------------------------------------------------
__global__ __launch_bounds__(256)
void combine_edges(const _Float16* __restrict__ a, const _Float16* __restrict__ c,
                   const int* __restrict__ nbr, _Float16* __restrict__ e)
{
    int idx = blockIdx.x * 256 + threadIdx.x;
    if (idx >= NB3 * 128) return;
    int ie = idx >> 7, ch = idx & 127;
    int node = ie / 3, k = ie - node * 3;
    int j = nbr[node * 3 + k];
    float v = (float)a[(size_t)node * 128 + ch] + (float)c[(size_t)j * 128 + ch];
    e[idx] = (_Float16)(v > 0.f ? v : 0.f);
}

__global__ __launch_bounds__(256)
void mean3(const _Float16* __restrict__ e, _Float16* __restrict__ d)
{
    int idx = blockIdx.x * 256 + threadIdx.x;
    if (idx >= NB * 128) return;
    int node = idx >> 7, ch = idx & 127;
    size_t base = (size_t)node * 3 * 128 + ch;
    float v = ((float)e[base] + (float)e[base + 128] + (float)e[base + 256]) * (1.f / 3.f);
    d[idx] = (_Float16)v;
}

// ---------------------------------------------------------------------------
// Heads: sigma (256-dot) and sigmoid RGB (128x3)
// ---------------------------------------------------------------------------
__global__ __launch_bounds__(256)
void final_out(const _Float16* __restrict__ hlast, const _Float16* __restrict__ d2b,
               const float* __restrict__ w_sigma, const float* __restrict__ b_sigma,
               const float* __restrict__ w_rgb, const float* __restrict__ b_rgb,
               float* __restrict__ out)
{
    int i = blockIdx.x * 256 + threadIdx.x;
    if (i >= NB) return;
    float s = b_sigma[0];
    for (int k = 0; k < 256; k++) s += (float)hlast[(size_t)i * 256 + k] * w_sigma[k];
    float a0 = b_rgb[0], a1 = b_rgb[1], a2 = b_rgb[2];
    for (int k = 0; k < 128; k++) {
        float dv = (float)d2b[(size_t)i * 128 + k];
        a0 += dv * w_rgb[k * 3 + 0];
        a1 += dv * w_rgb[k * 3 + 1];
        a2 += dv * w_rgb[k * 3 + 2];
    }
    out[(size_t)i * 4 + 0] = 1.f / (1.f + __expf(-a0));
    out[(size_t)i * 4 + 1] = 1.f / (1.f + __expf(-a1));
    out[(size_t)i * 4 + 2] = 1.f / (1.f + __expf(-a2));
    out[(size_t)i * 4 + 3] = s;
}

// ---------------------------------------------------------------------------
extern "C" void kernel_launch(void* const* d_in, const int* in_sizes, int n_in,
                              void* d_out, int out_size, void* d_ws, size_t ws_size,
                              hipStream_t stream)
{
    (void)in_sizes; (void)n_in; (void)out_size; (void)ws_size;
    const float* x       = (const float*)d_in[0];
    const int*   bid     = (const int*)  d_in[1];
    const float* w0      = (const float*)d_in[2];
    const float* b0      = (const float*)d_in[3];
    const float* w_mid   = (const float*)d_in[4];
    const float* b_mid   = (const float*)d_in[5];
    const float* w_skip  = (const float*)d_in[6];
    const float* b_skip  = (const float*)d_in[7];
    const float* w_final = (const float*)d_in[8];
    const float* b_final = (const float*)d_in[9];
    const float* w_sigma = (const float*)d_in[10];
    const float* b_sigma = (const float*)d_in[11];
    const float* e1_w1   = (const float*)d_in[12];
    const float* e1_b1   = (const float*)d_in[13];
    const float* e1_w2   = (const float*)d_in[14];
    const float* e1_b2   = (const float*)d_in[15];
    const float* e2_w1   = (const float*)d_in[16];
    const float* e2_b1   = (const float*)d_in[17];
    const float* e2_w2   = (const float*)d_in[18];
    const float* e2_b2   = (const float*)d_in[19];
    const float* w_rgb   = (const float*)d_in[20];
    const float* b_rgb   = (const float*)d_in[21];
    float* out = (float*)d_out;

    // -------- workspace layout --------
    char* ws = (char*)d_ws;
    size_t off = 0;
    auto alloc = [&](size_t bytes) -> char* {
        char* p = ws + off;
        off += (bytes + 255) & ~(size_t)255;
        return p;
    };
    _Float16* xyz16   = (_Float16*)alloc((size_t)NB * 64 * 2);
    _Float16* h_a     = (_Float16*)alloc((size_t)NB * 256 * 2);
    _Float16* h_b     = (_Float16*)alloc((size_t)NB * 256 * 2);
    _Float16* cat320  = (_Float16*)alloc((size_t)NB * 320 * 2);
    _Float16* featdir = (_Float16*)alloc((size_t)NB * 288 * 2);
    _Float16* abuf    = (_Float16*)alloc((size_t)NB * 128 * 2);   // reused for both convs
    _Float16* cbuf    = (_Float16*)alloc((size_t)NB * 128 * 2);   // reused
    _Float16* ebuf    = (_Float16*)alloc((size_t)NB3 * 128 * 2);  // reused
    _Float16* eobuf   = (_Float16*)alloc((size_t)NB3 * 128 * 2);  // reused
    _Float16* d1      = (_Float16*)alloc((size_t)NB * 128 * 2);
    _Float16* d2b     = (_Float16*)alloc((size_t)NB * 128 * 2);
    _Float16* w0p     = (_Float16*)alloc((size_t)256 * 64 * 2);   // [N][K] transposed
    _Float16* wmid16  = (_Float16*)alloc((size_t)6 * 256 * 256 * 2);
    _Float16* wskip16 = (_Float16*)alloc((size_t)256 * 320 * 2);
    _Float16* wfin16  = (_Float16*)alloc((size_t)256 * 256 * 2);
    _Float16* w1d     = (_Float16*)alloc((size_t)128 * 288 * 2);
    _Float16* w1b     = (_Float16*)alloc((size_t)128 * 288 * 2);
    _Float16* e1w2p   = (_Float16*)alloc((size_t)128 * 128 * 2);
    _Float16* e2d     = (_Float16*)alloc((size_t)128 * 128 * 2);
    _Float16* e2bw    = (_Float16*)alloc((size_t)128 * 128 * 2);
    _Float16* e2w2p   = (_Float16*)alloc((size_t)128 * 128 * 2);
    int* glist = (int*)alloc((size_t)NB * 4);
    int* gcnt  = (int*)alloc(32 * 4);
    int* nbr   = (int*)alloc((size_t)NB * 3 * 4);

    auto cdiv = [](int a, int b) { return (a + b - 1) / b; };

    // -------- staging --------
    zero_cnt<<<1, 32, 0, stream>>>(gcnt);
    build_groups<<<cdiv(NB, 256), 256, 0, stream>>>(bid, gcnt, glist);
    prep_kernel<<<cdiv(NB, 256), 256, 0, stream>>>(x, xyz16, cat320, featdir);

    conv_pad_t<<<cdiv(256 * 64, 256), 256, 0, stream>>>(w0, w0p, 63, 64, 256);
    for (int m = 0; m < 6; m++)
        conv_pad_t<<<cdiv(256 * 256, 256), 256, 0, stream>>>(
            w_mid + (size_t)m * 65536, wmid16 + (size_t)m * 65536, 256, 256, 256);
    conv_pad_t<<<cdiv(256 * 320, 256), 256, 0, stream>>>(w_skip, wskip16, 319, 320, 256);
    conv_pad_t<<<cdiv(256 * 256, 256), 256, 0, stream>>>(w_final, wfin16, 256, 256, 256);
    conv_pad_t<<<cdiv(128 * 128, 256), 256, 0, stream>>>(e1_w2, e1w2p, 128, 128, 128);
    conv_pad_t<<<cdiv(128 * 128, 256), 256, 0, stream>>>(e2_w2, e2w2p, 128, 128, 128);
    make_edge_w_t<<<cdiv(128 * 288, 256), 256, 0, stream>>>(e1_w1, w1d, w1b, 283, 288, 128);
    make_edge_w_t<<<cdiv(128 * 128, 256), 256, 0, stream>>>(e2_w1, e2d, e2bw, 128, 128, 128);

    // -------- kNN (VALU, ~11 GF) --------
    knn_kernel<<<NG * 8, 256, 0, stream>>>(x, glist, nbr);

    // -------- trunk MLP (WMMA) --------
    const dim3 gN256(256 / 128, NB / 128);   // (2, 320)
    const dim3 gN128(1, NB / 128);           // (1, 320)
    const dim3 gE128(1, NB3 / 128);          // (1, 960)

    wmma_gemm<true, true><<<gN256, 256, 0, stream>>>(xyz16, w0p, b0, h_a, 64, 256, 256);          // L0
    wmma_gemm<true, true><<<gN256, 256, 0, stream>>>(h_a, wmid16 + 0 * 65536, b_mid + 0 * 256, h_b, 256, 256, 256); // L1
    wmma_gemm<true, true><<<gN256, 256, 0, stream>>>(h_b, wmid16 + 1 * 65536, b_mid + 1 * 256, h_a, 256, 256, 256); // L2
    wmma_gemm<true, true><<<gN256, 256, 0, stream>>>(h_a, wmid16 + 2 * 65536, b_mid + 2 * 256, cat320 + 63, 256, 256, 320); // L3 -> concat slot
    wmma_gemm<true, true><<<gN256, 256, 0, stream>>>(cat320, wskip16, b_skip, h_a, 320, 256, 256); // L4 skip
    wmma_gemm<true, true><<<gN256, 256, 0, stream>>>(h_a, wmid16 + 3 * 65536, b_mid + 3 * 256, h_b, 256, 256, 256); // L5
    wmma_gemm<true, true><<<gN256, 256, 0, stream>>>(h_b, wmid16 + 4 * 65536, b_mid + 4 * 256, h_a, 256, 256, 256); // L6
    wmma_gemm<true, true><<<gN256, 256, 0, stream>>>(h_a, wmid16 + 5 * 65536, b_mid + 5 * 256, h_b, 256, 256, 256); // L7 (h_last)
    wmma_gemm<false, true><<<gN256, 256, 0, stream>>>(h_b, wfin16, b_final, featdir, 256, 256, 288); // feat (no relu) -> [feat,dir]

    // -------- EdgeConv 1 (WMMA + gather) --------
    wmma_gemm<false, true ><<<gN128, 256, 0, stream>>>(featdir, w1d, e1_b1, abuf, 288, 128, 128); // a_i = xi@(Wt-Wb)+b1
    wmma_gemm<false, false><<<gN128, 256, 0, stream>>>(featdir, w1b, nullptr, cbuf, 288, 128, 128); // c_j = xj@Wb
    combine_edges<<<cdiv(NB3 * 128, 256), 256, 0, stream>>>(abuf, cbuf, nbr, ebuf);
    wmma_gemm<true, true><<<gE128, 256, 0, stream>>>(ebuf, e1w2p, e1_b2, eobuf, 128, 128, 128);
    mean3<<<cdiv(NB * 128, 256), 256, 0, stream>>>(eobuf, d1);

    // -------- EdgeConv 2 (WMMA + gather) --------
    wmma_gemm<false, true ><<<gN128, 256, 0, stream>>>(d1, e2d, e2_b1, abuf, 128, 128, 128);
    wmma_gemm<false, false><<<gN128, 256, 0, stream>>>(d1, e2bw, nullptr, cbuf, 128, 128, 128);
    combine_edges<<<cdiv(NB3 * 128, 256), 256, 0, stream>>>(abuf, cbuf, nbr, ebuf);
    wmma_gemm<true, true><<<gE128, 256, 0, stream>>>(ebuf, e2w2p, e2_b2, eobuf, 128, 128, 128);
    mean3<<<cdiv(NB * 128, 256), 256, 0, stream>>>(eobuf, d2b);

    // -------- heads --------
    final_out<<<cdiv(NB, 256), 256, 0, stream>>>(h_b, d2b, w_sigma, b_sigma, w_rgb, b_rgb, out);
}
